// GCN_86818468921562
// MI455X (gfx1250) — matrix operand order repaired and checked by reading, not compiled, not run on hardware
//
#include <hip/hip_runtime.h>
#include <hip/hip_bf16.h>

#define KN   50000
#define KE   800000
#define KIN  144
#define KKS  16
#define KHID 256
#define KOUT 40

typedef __bf16 bh;
typedef __attribute__((ext_vector_type(16))) __bf16 v16bf;
typedef __attribute__((ext_vector_type(8)))  __bf16 v8bf;
typedef __attribute__((ext_vector_type(8)))  float  v8f;
typedef __attribute__((ext_vector_type(4)))  int    v4i;

#define AS1 __attribute__((address_space(1)))
#define AS3 __attribute__((address_space(3)))

#if __has_builtin(__builtin_amdgcn_global_load_async_to_lds_b128) && \
    __has_builtin(__builtin_amdgcn_s_wait_asynccnt)
#define HAVE_ASYNC_LDS 1
#else
#define HAVE_ASYNC_LDS 0
#endif

// ---------------------------------------------------------------------------
// WMMA GEMM:  out[n, m] = act( sum_k A[n,k] * W[m,k] + bias[m] )
// Block = 8 waves = 128 nodes x 64 features; the 64 x Kdim weight tile is
// staged once per block into LDS (async-to-LDS when available), inner loop:
// A fragments from global (2x b128/lane), B fragments from LDS (ds_load_b128).
// ---------------------------------------------------------------------------
__global__ __launch_bounds__(256)
void gcn_wmma_gemm(const bh* __restrict__ A, const bh* __restrict__ W,
                   const float* __restrict__ bias,
                   float* __restrict__ outF, bh* __restrict__ outB,
                   int rows, int ldA, int Kdim, int mTiles,
                   int Mvalid, int ldOut, int relu) {
  __shared__ __align__(16) bh ldsW[64 * 384];   // max tile: 64 feats x Kdim=384

  const int wave = threadIdx.x >> 5;
  const int lane = threadIdx.x & 31;
  const int half = lane >> 4;        // 0: lanes 0-15, 1: lanes 16-31
  const int lq   = lane & 15;

  const int mt            = (int)(blockIdx.x % mTiles);
  const long long rowBlk  = blockIdx.x / mTiles;
  const int mbase         = mt << 6;

  // ---- stage weight tile (rows are contiguous in W => flat 16B-chunk copy) ----
  const bh* Wt = W + (size_t)mbase * (size_t)Kdim;
  const int chunks = (64 * Kdim) >> 3;          // # of 16-byte chunks
#if HAVE_ASYNC_LDS
  for (int i = threadIdx.x; i < chunks; i += 256) {
    __builtin_amdgcn_global_load_async_to_lds_b128(
        (AS1 v4i*)(Wt + (i << 3)),
        (AS3 v4i*)(ldsW + (i << 3)), 0, 0);
  }
  __builtin_amdgcn_s_wait_asynccnt(0);
#else
  for (int i = threadIdx.x; i < chunks; i += 256)
    ((uint4*)ldsW)[i] = ((const uint4*)Wt)[i];
#endif
  __syncthreads();   // all waves (incl. tail-idle ones) reach the barrier

  const long long nodeBase = (rowBlk * 8 + wave) << 4;
  if (nodeBase < rows) {
    const bh* Arow = A + (size_t)(nodeBase + lq) * (size_t)ldA;

    v8f acc0 = {}, acc1 = {}, acc2 = {}, acc3 = {};

    for (int k0 = 0; k0 < Kdim; k0 += 32) {
      // ---- A fragment: 16x32 bf16; per lane = two contiguous 16B runs ----
      v8bf a0 = *(const v8bf*)(Arow + k0 + (half << 3));
      v8bf a1 = *(const v8bf*)(Arow + k0 + 16 + (half << 3));
      v16bf a = __builtin_shufflevector(a0, a1, 0, 1, 2, 3, 4, 5, 6, 7,
                                        8, 9, 10, 11, 12, 13, 14, 15);
      // ---- 4 B fragments from LDS: lane = feature col, two 16B runs (same k order as A) ----
#pragma unroll
      for (int t = 0; t < 4; ++t) {
        const bh* brow = ldsW + (size_t)((t << 4) + lq) * (size_t)Kdim
                              + k0 + (half << 3);
        v8bf b0 = *(const v8bf*)(brow);
        v8bf b1 = *(const v8bf*)(brow + 16);
        v16bf b = __builtin_shufflevector(b0, b1, 0, 1, 2, 3, 4, 5, 6, 7,
                                          8, 9, 10, 11, 12, 13, 14, 15);
        if (t == 0) acc0 = __builtin_amdgcn_wmma_f32_16x16x32_bf16(false, a, false, b, (short)0, acc0, false, false);
        if (t == 1) acc1 = __builtin_amdgcn_wmma_f32_16x16x32_bf16(false, a, false, b, (short)0, acc1, false, false);
        if (t == 2) acc2 = __builtin_amdgcn_wmma_f32_16x16x32_bf16(false, a, false, b, (short)0, acc2, false, false);
        if (t == 3) acc3 = __builtin_amdgcn_wmma_f32_16x16x32_bf16(false, a, false, b, (short)0, acc3, false, false);
      }
    }

    // ---- writeback: D layout lane = column (feature), VGPR v = row (node) ----
#pragma unroll
    for (int t = 0; t < 4; ++t) {
      const int feat = mbase + (t << 4) + lq;
      const v8f acc = (t == 0) ? acc0 : (t == 1) ? acc1 : (t == 2) ? acc2 : acc3;
      if (feat >= Mvalid) continue;
      const float bv = bias ? bias[feat] : 0.0f;
#pragma unroll
      for (int v = 0; v < 8; ++v) {
        const long long node = nodeBase + v + (half << 3);
        if (node >= rows) continue;
        float val = acc[v] + bv;
        if (relu) val = fmaxf(val, 0.0f);
        if (outF) outF[(size_t)node * ldOut + feat] = val;
        if (outB) outB[(size_t)node * ldOut + feat] = (bh)val;
      }
    }
  }
}

// ---------------------------------------------------------------------------
// f32 -> bf16 weight conversion with zero padding (rows and cols)
// ---------------------------------------------------------------------------
__global__ void k_cvt(const float* __restrict__ src, bh* __restrict__ dst,
                      int drows, int dcols, int srows, int scols) {
  long long i = (long long)blockIdx.x * blockDim.x + threadIdx.x;
  long long tot = (long long)drows * dcols;
  if (i >= tot) return;
  int r = (int)(i / dcols), c = (int)(i % dcols);
  float v = (r < srows && c < scols) ? src[(size_t)r * scols + c] : 0.0f;
  dst[i] = (bh)v;
}

// Build [xs ; -xs] as 2N x 32 bf16 (cols 16..31 zero padded); xs = x[:, 128:144]
__global__ void k_build_xspm(const float* __restrict__ x, bh* __restrict__ dst, int n) {
  long long i = (long long)blockIdx.x * blockDim.x + threadIdx.x;
  long long tot = (long long)2 * n * 32;
  if (i >= tot) return;
  long long r2 = i >> 5; int j = (int)(i & 31);
  long long node = (r2 < n) ? r2 : (r2 - n);
  float s = (r2 < n) ? 1.0f : -1.0f;
  float v = (j < KKS) ? s * x[node * KIN + (KIN - KKS) + j] : 0.0f;
  dst[i] = (bh)v;
}

// h0 = concat( x[:, :128], Hp + Hm )  as N x 384 bf16
__global__ void k_mlp_combine(const float* __restrict__ Hp, const float* __restrict__ Hm,
                              const float* __restrict__ x, bh* __restrict__ h0, int n) {
  long long i = (long long)blockIdx.x * blockDim.x + threadIdx.x;
  long long tot = (long long)n * 384;
  if (i >= tot) return;
  long long node = i / 384; int c = (int)(i % 384);
  float v;
  if (c < KIN - KKS) v = x[node * KIN + c];
  else {
    int cc = c - (KIN - KKS);
    v = Hp[node * KHID + cc] + Hm[node * KHID + cc];
  }
  h0[i] = (bh)v;
}

__global__ void k_zero_f32(float* __restrict__ p, long long n) {
  long long i = (long long)blockIdx.x * blockDim.x + threadIdx.x;
  if (i < n) p[i] = 0.0f;
}

__global__ void k_deg_count(const long long* __restrict__ ei, float* __restrict__ deg, int e) {
  long long i = (long long)blockIdx.x * blockDim.x + threadIdx.x;
  if (i >= e) return;
  long long c = ei[(long long)e + i];
  unsafeAtomicAdd(&deg[c], 1.0f);
}

// deg -> dinv, including self loop (+1); deg>=1 always so dinv = rsqrt(deg)
__global__ void k_deg_finish(float* __restrict__ deg, int n) {
  long long i = (long long)blockIdx.x * blockDim.x + threadIdx.x;
  if (i >= n) return;
  float d = deg[i] + 1.0f;
  deg[i] = rsqrtf(fmaxf(d, 1.0f));
}

__global__ void k_edge_norm(const long long* __restrict__ ei, const float* __restrict__ dinv,
                            float* __restrict__ nrm, int e) {
  long long i = (long long)blockIdx.x * blockDim.x + threadIdx.x;
  if (i >= e) return;
  long long r = ei[i], c = ei[(long long)e + i];
  nrm[i] = dinv[r] * dinv[c];
}

// one wave per edge; 32 lanes x 8 coalesced f32 atomics into agg[col]
__global__ __launch_bounds__(256)
void k_agg_edges(const long long* __restrict__ ei, const float* __restrict__ nrm,
                 const float* __restrict__ H, float* __restrict__ agg, int e) {
  const int wave = threadIdx.x >> 5;
  const int lane = threadIdx.x & 31;
  long long edge = (long long)blockIdx.x * 8 + wave;
  if (edge >= e) return;
  long long r = ei[edge], c = ei[(long long)e + edge];
  float nv = nrm[edge];
  const float* src = H + r * KHID;
  float* dst = agg + c * KHID;
#pragma unroll
  for (int i = 0; i < 8; ++i) {
    int f = lane + (i << 5);
    unsafeAtomicAdd(&dst[f], src[f] * nv);
  }
}

// out = relu(agg + H*dinv^2 (self loop) + bias); bf16 for next layer, optional f32
__global__ void k_layer_finish(const float* __restrict__ agg, const float* __restrict__ H,
                               const float* __restrict__ dinv, const float* __restrict__ bias,
                               bh* __restrict__ outB, float* __restrict__ outF, int n) {
  long long i = (long long)blockIdx.x * blockDim.x + threadIdx.x;
  long long tot = (long long)n * KHID;
  if (i >= tot) return;
  long long node = i / KHID; int c = (int)(i % KHID);
  float di = dinv[node];
  float v = agg[i] + H[i] * di * di + bias[c];
  v = fmaxf(v, 0.0f);
  outB[i] = (bh)v;
  if (outF) outF[i] = v;
}

// ---------------------------------------------------------------------------
extern "C" void kernel_launch(void* const* d_in, const int* in_sizes, int n_in,
                              void* d_out, int out_size, void* d_ws, size_t ws_size,
                              hipStream_t stream) {
  const float*     x     = (const float*)d_in[0];
  const long long* ei    = (const long long*)d_in[1];
  const float*     sn_w1 = (const float*)d_in[2];
  const float*     sn_b1 = (const float*)d_in[3];
  const float*     sn_w2 = (const float*)d_in[4];
  const float*     sn_b2 = (const float*)d_in[5];
  const float*     w1    = (const float*)d_in[6];
  const float*     b1    = (const float*)d_in[7];
  const float*     w2    = (const float*)d_in[8];
  const float*     b2    = (const float*)d_in[9];
  const float*     w3    = (const float*)d_in[10];
  const float*     b3    = (const float*)d_in[11];
  const float*     w4    = (const float*)d_in[12];
  const float*     b4    = (const float*)d_in[13];
  const float*     wo    = (const float*)d_in[14];
  const float*     bo    = (const float*)d_in[15];

  float* out_h = (float*)d_out;                       // N x 256
  float* out_z = out_h + (size_t)KN * KHID;           // N x 40

  // -------- workspace layout --------
  char* ws = (char*)d_ws;
  size_t off = 0;
  auto alloc = [&](size_t bytes) -> char* {
    char* p = ws + off;
    off += (bytes + 255) & ~(size_t)255;
    return p;
  };
  bh* W_sn1 = (bh*)alloc((size_t)KHID * 32 * 2);
  bh* W_sn2 = (bh*)alloc((size_t)KHID * KHID * 2);
  bh* W_w1  = (bh*)alloc((size_t)KHID * 384 * 2);
  bh* W_w2  = (bh*)alloc((size_t)KHID * KHID * 2);
  bh* W_w3  = (bh*)alloc((size_t)KHID * KHID * 2);
  bh* W_w4  = (bh*)alloc((size_t)KHID * KHID * 2);
  bh* W_wo  = (bh*)alloc((size_t)64 * KHID * 2);
  float* dinv = (float*)alloc((size_t)KN * 4);
  float* enrm = (float*)alloc((size_t)KE * 4);
  float* P0   = (float*)alloc((size_t)KN * KHID * 4);        // GEMM f32 out
  float* P2   = (float*)alloc((size_t)KN * KHID * 4);        // agg
  bh*    P1   = (bh*)alloc((size_t)2 * KN * KHID * 2);       // bf16 features
  bh*    P3   = (bh*)alloc((size_t)2 * KN * 32 * 2);         // [xs;-xs]

  const int T = 256;
  auto nb = [](long long n, int t) { return (unsigned)((n + t - 1) / t); };
  // GEMM grid: ceil(rowTiles/8) row-blocks x mTiles feature-tiles
  auto gemm_blocks = [](long long rows, int mTiles) {
    long long rowTiles = (rows + 15) / 16;
    return (unsigned)(((rowTiles + 7) / 8) * mTiles);
  };

  // -------- weights -> bf16 (padded) --------
  k_cvt<<<nb((long long)KHID * 32, T), T, 0, stream>>>(sn_w1, W_sn1, KHID, 32, KHID, KKS);
  k_cvt<<<nb((long long)KHID * KHID, T), T, 0, stream>>>(sn_w2, W_sn2, KHID, KHID, KHID, KHID);
  k_cvt<<<nb((long long)KHID * 384, T), T, 0, stream>>>(w1, W_w1, KHID, 384, KHID, 384);
  k_cvt<<<nb((long long)KHID * KHID, T), T, 0, stream>>>(w2, W_w2, KHID, KHID, KHID, KHID);
  k_cvt<<<nb((long long)KHID * KHID, T), T, 0, stream>>>(w3, W_w3, KHID, KHID, KHID, KHID);
  k_cvt<<<nb((long long)KHID * KHID, T), T, 0, stream>>>(w4, W_w4, KHID, KHID, KHID, KHID);
  k_cvt<<<nb((long long)64 * KHID, T), T, 0, stream>>>(wo, W_wo, 64, KHID, KOUT, KHID);

  // -------- spectral MLP on [xs; -xs] --------
  k_build_xspm<<<nb((long long)2 * KN * 32, T), T, 0, stream>>>(x, P3, KN);

  gcn_wmma_gemm<<<gemm_blocks(2 * KN, 4), 256, 0, stream>>>(P3, W_sn1, sn_b1, nullptr, P1,
      2 * KN, 32, 32, 4, KHID, KHID, 1);
  gcn_wmma_gemm<<<gemm_blocks(KN, 4), 256, 0, stream>>>(P1, W_sn2, sn_b2, P0, nullptr,
      KN, KHID, KHID, 4, KHID, KHID, 1);
  gcn_wmma_gemm<<<gemm_blocks(KN, 4), 256, 0, stream>>>(P1 + (size_t)KN * KHID, W_sn2, sn_b2, P2, nullptr,
      KN, KHID, KHID, 4, KHID, KHID, 1);

  k_mlp_combine<<<nb((long long)KN * 384, T), T, 0, stream>>>(P0, P2, x, P1, KN);

  // -------- GCN normalization --------
  k_zero_f32<<<nb(KN, T), T, 0, stream>>>(dinv, KN);
  k_deg_count<<<nb(KE, T), T, 0, stream>>>(ei, dinv, KE);
  k_deg_finish<<<nb(KN, T), T, 0, stream>>>(dinv, KN);
  k_edge_norm<<<nb(KE, T), T, 0, stream>>>(ei, dinv, enrm, KE);

  // -------- 4 GCN layers --------
  const bh*    Wl[4] = {W_w1, W_w2, W_w3, W_w4};
  const float* Bl[4] = {b1, b2, b3, b4};
  const int    Kl[4] = {384, KHID, KHID, KHID};
  for (int l = 0; l < 4; ++l) {
    gcn_wmma_gemm<<<gemm_blocks(KN, 4), 256, 0, stream>>>(P1, Wl[l], nullptr, P0, nullptr,
        KN, Kl[l], Kl[l], 4, KHID, KHID, 0);
    k_zero_f32<<<nb((long long)KN * KHID, T), T, 0, stream>>>(P2, (long long)KN * KHID);
    k_agg_edges<<<nb(KE, 8), 256, 0, stream>>>(ei, enrm, P0, P2, KE);
    k_layer_finish<<<nb((long long)KN * KHID, T), T, 0, stream>>>(
        P2, P0, dinv, Bl[l], P1, (l == 3) ? out_h : nullptr, KN);
  }

  // -------- output head z = h @ wo.T + bo --------
  gcn_wmma_gemm<<<gemm_blocks(KN, 1), 256, 0, stream>>>(P1, W_wo, bo, out_z, nullptr,
      KN, KHID, KHID, 1, KOUT, KOUT, 0);
}
